// AGNNConvDGL_66812511256932
// MI455X (gfx1250) — matrix operand order repaired and checked by reading, not compiled, not run on hardware
//
#include <hip/hip_runtime.h>
#include <hip/hip_bf16.h>
#include <stdint.h>

typedef __attribute__((ext_vector_type(16))) _Float16 v16h;
typedef __attribute__((ext_vector_type(8)))  _Float16 v8h;
typedef __attribute__((ext_vector_type(8)))  float    v8f;

#define EPSN 1e-12f

__device__ __forceinline__ float waveReduceSum(float v) {
#pragma unroll
  for (int m = 16; m >= 1; m >>= 1) v += __shfl_xor(v, m, 32);
  return v;
}
__device__ __forceinline__ float waveReduceMax(float v) {
#pragma unroll
  for (int m = 16; m >= 1; m >>= 1) v = fmaxf(v, __shfl_xor(v, m, 32));
  return v;
}

// Kernel 1: per-node inverse L2 norm + f16 copy of H (WMMA A operand source).
__global__ void agnn_norm_kernel(const float* __restrict__ H,
                                 float* __restrict__ invn,
                                 _Float16* __restrict__ Hh, int Nn) {
  int wave = threadIdx.x >> 5, lane = threadIdx.x & 31;
  int node = blockIdx.x * (blockDim.x >> 5) + wave;
  if (node >= Nn) return;
  const float2* h2 = reinterpret_cast<const float2*>(H) + (size_t)node * 32;
  float2 v = h2[lane];
  float s = waveReduceSum(v.x * v.x + v.y * v.y);
  float inv = 1.0f / fmaxf(sqrtf(s), EPSN);
  if (lane == 0) invn[node] = inv;
  _Float16* dst = Hh + (size_t)node * 64 + lane * 2;
  dst[0] = (_Float16)v.x;
  dst[1] = (_Float16)v.y;
}

// Kernel 2: CSR row offsets from the sorted int64 dst array (lower_bound).
__global__ void agnn_rowptr_kernel(const long long* __restrict__ row,
                                   int* __restrict__ rowptr, int Nn, int Ee) {
  int t = blockIdx.x * blockDim.x + threadIdx.x;
  if (t > Nn) return;
  int lo = 0, hi = Ee;
  while (lo < hi) {
    int mid = (lo + hi) >> 1;
    if (row[mid] < (long long)t) lo = mid + 1; else hi = mid;
  }
  rowptr[t] = lo;
}

// Kernel 3: SDDMM — one thread per edge, float4 dot in f32 (softmax input stays accurate).
__global__ void agnn_sddmm_kernel(const float* __restrict__ H,
                                  const long long* __restrict__ row,
                                  const long long* __restrict__ col,
                                  const float* __restrict__ invn,
                                  float* __restrict__ scores, int Ee) {
  int e = blockIdx.x * blockDim.x + threadIdx.x;
  if (e >= Ee) return;
  long long r = row[e], c = col[e];
  const float4* a = reinterpret_cast<const float4*>(H) + r * 16;
  const float4* b = reinterpret_cast<const float4*>(H) + c * 16;
  float s = 0.f;
#pragma unroll
  for (int i = 0; i < 16; ++i) {
    float4 x = a[i], y = b[i];
    s += x.x * y.x + x.y * y.y + x.z * y.z + x.w * y.w;
  }
  scores[e] = s * invn[r] * invn[c];
}

// Kernel 4: segment softmax per destination node; one wave per node.
__global__ void agnn_softmax_kernel(const float* __restrict__ scores,
                                    const int* __restrict__ rowptr,
                                    _Float16* __restrict__ attnh, int Nn) {
  int wave = threadIdx.x >> 5, lane = threadIdx.x & 31;
  int node = blockIdx.x * (blockDim.x >> 5) + wave;
  if (node >= Nn) return;
  int s0 = rowptr[node], s1 = rowptr[node + 1];
  float m = -__builtin_inff();
  for (int e = s0 + lane; e < s1; e += 32) m = fmaxf(m, scores[e]);
  m = waveReduceMax(m);
  float sum = 0.f;
  for (int e = s0 + lane; e < s1; e += 32) sum += __expf(scores[e] - m);
  sum = waveReduceSum(sum);
  float rinv = (sum > 0.f) ? (1.0f / sum) : 0.f;
  for (int e = s0 + lane; e < s1; e += 32)
    attnh[e] = (_Float16)(__expf(scores[e] - m) * rinv);
}

// Kernel 5: row-window SpMM via v_wmma_f32_16x16x32_f16, operand roles swapped:
//   A (16x32 f16) = features of the 32 chunk edges (from transposed LDS tile, 2xb128/lane)
//   B (32x16 f16) = attn one-hot per destination (2 aligned 16B loads + cndmask masks)
//   D (16x16 f32) = out[dest][feature-tile] -> contiguous float4 stores.
#define KP 40  // padded LDS row stride in halves (80 B: 16B-aligned, conflict-free)

__global__ void agnn_spmm_wmma_kernel(const _Float16* __restrict__ Hh,
                                      const long long* __restrict__ col,
                                      const _Float16* __restrict__ attnh,
                                      const int* __restrict__ rowptr,
                                      float* __restrict__ out, int Nn) {
  __shared__ _Float16 T[64 * KP];     // transposed tile: T[f*KP + k], 5 KB
  int lane = threadIdx.x & 31;
  int base = blockIdx.x * 16;
  if (base >= Nn) return;

  int S    = rowptr[base];
  int Eend = rowptr[min(base + 16, Nn)];
  int S0   = S & ~31;                 // 32-aligned chunk start; leading edges mask to 0
  int r    = lane & 15;               // destination column (B: N dim) / feature row (A: M dim)
  int rs   = rowptr[min(base + r, Nn)];
  int re   = rowptr[min(base + r + 1, Nn)];
  int kbA  = (lane < 16) ? 0 : 16;    // A halves: K = kbA+{0..15}
  int kbB  = (lane < 16) ? 0 : 16;    // B halves: K = kbB+{0..15}

  v8f acc0 = {}, acc1 = {}, acc2 = {}, acc3 = {};

  for (int off = S0; off < Eend; off += 32) {
    // ---- stage this lane's edge feature row, transposed, into LDS ----
    int e = off + lane;
    long long c = (e < Eend) ? col[e] : 0;
    const uint4* src = reinterpret_cast<const uint4*>(Hh) + c * 8;
    uint4 rrow[8];
    uint4 z; z.x = z.y = z.z = z.w = 0u;
#pragma unroll
    for (int j = 0; j < 8; ++j) rrow[j] = (e < Eend) ? src[j] : z;  // zero-fill: no NaN*0
    const _Float16* hs = reinterpret_cast<const _Float16*>(rrow);
#pragma unroll
    for (int f = 0; f < 64; ++f) T[f * KP + lane] = hs[f];          // conflict-free b16 scatter
    __syncthreads();

    // ---- B fragment: attn values masked by CSR ownership of destination r ----
    const v8h* ap = reinterpret_cast<const v8h*>(attnh + off + kbB); // 32B-aligned (off%32==0)
    v8h blo = ap[0], bhi = ap[1];
    v16h bv = __builtin_shufflevector(blo, bhi, 0, 1, 2, 3, 4, 5, 6, 7,
                                      8, 9, 10, 11, 12, 13, 14, 15);
    v16h b;
#pragma unroll
    for (int i = 0; i < 16; ++i) {
      int e2 = off + kbB + i;
      bool own = (e2 >= rs) && (e2 < re);
      b[i] = own ? bv[i] : (_Float16)0.0f;
    }

    // ---- A fragments (one per 16-feature tile) straight from transposed LDS ----
#pragma unroll
    for (int t = 0; t < 4; ++t) {
      const _Float16* Trow = T + (t * 16 + r) * KP;
      v8h alo = *reinterpret_cast<const v8h*>(Trow + kbA);        // K kbA..kbA+7
      v8h ahi = *reinterpret_cast<const v8h*>(Trow + kbA + 8);    // K kbA+8..kbA+15
      v16h a = __builtin_shufflevector(alo, ahi, 0, 1, 2, 3, 4, 5, 6, 7,
                                       8, 9, 10, 11, 12, 13, 14, 15);
      if (t == 0) acc0 = __builtin_amdgcn_wmma_f32_16x16x32_f16(false, a, false, b, (short)0, acc0, false, false);
      if (t == 1) acc1 = __builtin_amdgcn_wmma_f32_16x16x32_f16(false, a, false, b, (short)0, acc1, false, false);
      if (t == 2) acc2 = __builtin_amdgcn_wmma_f32_16x16x32_f16(false, a, false, b, (short)0, acc2, false, false);
      if (t == 3) acc3 = __builtin_amdgcn_wmma_f32_16x16x32_f16(false, a, false, b, (short)0, acc3, false, false);
    }
    __syncthreads();
  }

  // ---- store: D[f][n] -> out[base+n][ftile*16 + f]; 8 contiguous floats per lane/tile ----
  int row_out = base + r;
  if (row_out < Nn) {
    float* o = out + (size_t)row_out * 64 + ((lane < 16) ? 0 : 8);
    const float4* a0 = reinterpret_cast<const float4*>(&acc0);
    const float4* a1 = reinterpret_cast<const float4*>(&acc1);
    const float4* a2 = reinterpret_cast<const float4*>(&acc2);
    const float4* a3 = reinterpret_cast<const float4*>(&acc3);
    reinterpret_cast<float4*>(o +  0)[0] = a0[0];
    reinterpret_cast<float4*>(o +  4)[0] = a0[1];
    reinterpret_cast<float4*>(o + 16)[0] = a1[0];
    reinterpret_cast<float4*>(o + 20)[0] = a1[1];
    reinterpret_cast<float4*>(o + 32)[0] = a2[0];
    reinterpret_cast<float4*>(o + 36)[0] = a2[1];
    reinterpret_cast<float4*>(o + 48)[0] = a3[0];
    reinterpret_cast<float4*>(o + 52)[0] = a3[1];
  }
}

extern "C" void kernel_launch(void* const* d_in, const int* in_sizes, int n_in,
                              void* d_out, int out_size, void* d_ws, size_t ws_size,
                              hipStream_t stream) {
  const float*     H   = (const float*)d_in[0];
  const long long* row = (const long long*)d_in[1];
  const long long* col = (const long long*)d_in[2];
  (void)n_in; (void)out_size; (void)ws_size;

  const int D  = 64;
  int Nn = in_sizes[0] / D;
  int Ee = in_sizes[1];
  float* out = (float*)d_out;

  // Workspace carve-out (256 B aligned): ~23 MB total.
  char* ws = (char*)d_ws;
  size_t off = 0;
  auto take = [&](size_t bytes) -> void* {
    void* p = ws + off;
    off = (off + bytes + 255) & ~(size_t)255;
    return p;
  };
  float*    invn   = (float*)take(sizeof(float) * (size_t)Nn);
  _Float16* Hh     = (_Float16*)take(sizeof(_Float16) * (size_t)Nn * D);
  int*      rowptr = (int*)take(sizeof(int) * (size_t)(Nn + 1));
  float*    scores = (float*)take(sizeof(float) * (size_t)Ee);
  _Float16* attnh  = (_Float16*)take(sizeof(_Float16) * (size_t)Ee + 64); // slack for aligned tail read

  agnn_norm_kernel<<<(Nn + 7) / 8, 256, 0, stream>>>(H, invn, Hh, Nn);
  agnn_rowptr_kernel<<<(Nn + 1 + 255) / 256, 256, 0, stream>>>(row, rowptr, Nn, Ee);
  agnn_sddmm_kernel<<<(Ee + 255) / 256, 256, 0, stream>>>(H, row, col, invn, scores, Ee);
  agnn_softmax_kernel<<<(Nn + 7) / 8, 256, 0, stream>>>(scores, rowptr, attnh, Nn);

  int ntiles = (Nn + 15) / 16;
  agnn_spmm_wmma_kernel<<<ntiles, 32, 0, stream>>>(Hh, col, attnh, rowptr, out, Nn);
}
